// DistanceEstimator_68101001445685
// MI455X (gfx1250) — matrix operand order repaired and checked by reading, not compiled, not run on hardware
//
#include <hip/hip_runtime.h>
#include <hip/hip_bf16.h>

typedef __attribute__((ext_vector_type(16))) _Float16 v16h;
typedef __attribute__((ext_vector_type(4)))  _Float16 v4h;
typedef __attribute__((ext_vector_type(8)))  float    v8f;
typedef __attribute__((ext_vector_type(4)))  float    v4f;

#define WMMA_F16(a, b, c) \
  __builtin_amdgcn_wmma_f32_16x16x32_f16(false, (a), false, (b), (short)0, (c), false, false)

// Intra-wave LDS produce->consume ordering (each wave owns a private LDS
// slice; DScnt==0 makes the wave's DS stores visible to all its lanes).
#define LDS_FENCE() asm volatile("s_wait_dscnt 0" ::: "memory")

// ---- WMMA fragment loaders (ISA 7.12.2 layouts) ----------------------------
// A: 16x32 f16 from row-major LDS, leading dim ld.  lane&15 = M.
// element e -> K = kb + (e<8 ? e : e+8) + 8*(lane>=16); two contiguous 16B runs
__device__ __forceinline__ v16h lda_f16(const _Float16* m, int ld, int kb, int lane) {
  const int g = lane >> 4, r = lane & 15;
  v16h a;
#pragma unroll
  for (int e = 0; e < 16; ++e) {
    const int k = kb + ((e < 8) ? e : (e + 8)) + g * 8;
    a[e] = m[r * ld + k];
  }
  return a;
}
// B: fragment-packed weights — lane's 16 halves contiguous (32B, aligned).
// Packed layout produced by pack_b_kernel below.
__device__ __forceinline__ v16h ldb_packed(const _Float16* wp, int numCT, int kc, int ct,
                                           int lane) {
  return *(const v16h*)(wp + (((size_t)(kc * numCT + ct) * 32 + lane) << 4));
}

// Pack row-major W[K][Ncols] f32 into per-fragment B layout f16:
// dst[(((kc*numCT+ct)*32+lane)*16)+e] = W[kc*32 + 8*(lane>=16) + (e<8 ? e : e+8)][ct*16 + (lane&15)]
__global__ void pack_b_kernel(const float* __restrict__ w, _Float16* __restrict__ dst,
                              int K, int Ncols) {
  const int numCT = Ncols >> 4;
  const int o = blockIdx.x * blockDim.x + threadIdx.x;
  if (o >= K * Ncols) return;
  const int e = o & 15;
  const int lane = (o >> 4) & 31;
  const int rest = o >> 9;
  const int ct = rest % numCT;
  const int kc = rest / numCT;
  const int k = kc * 32 + (lane >> 4) * 8 + ((e < 8) ? e : (e + 8));
  const int c = ct * 16 + (lane & 15);
  dst[o] = (_Float16)w[k * Ncols + c];
}

__global__ void zero_kernel(float* __restrict__ p, long n) {
  long i = (long)blockIdx.x * blockDim.x + threadIdx.x;
  if (i < n) p[i] = 0.0f;
}

// ---- node id-MLP: x0[n] = mlp2(clip(name/2^48,0,1))  (1->64->64) -----------
__global__ void node_encode_kernel(const int* __restrict__ names,
                                   const float* __restrict__ w1, const float* __restrict__ b1,
                                   const _Float16* __restrict__ w2p, const float* __restrict__ b2,
                                   float* __restrict__ x0, int N) {
  const int wave = threadIdx.x >> 5, lane = threadIdx.x & 31;
  const int tile = blockIdx.x * 4 + wave;
  const int numTiles = (N + 15) >> 4;
  __shared__ alignas(16) _Float16 hbuf[4][16 * 64];
  if (tile >= numTiles) return;
  const int rbase = tile * 16;
  const bool full = (rbase + 16 <= N);
  // hidden h = relu(norm * w1 + b1) : 16x64 f16, 4-wide per lane
#pragma unroll
  for (int t = 0; t < 8; ++t) {
    const int idx4 = (lane + 32 * t) * 4;
    const int r = idx4 >> 6, c = idx4 & 63;
    const int gr = rbase + r;
    float nm = 0.0f;
    if (full || gr < N) {
      float v = (float)names[gr] * (1.0f / 281474976710655.0f);
      nm = fminf(fmaxf(v, 0.0f), 1.0f);
    }
    const v4f wv = *(const v4f*)(w1 + c);
    const v4f bv = *(const v4f*)(b1 + c);
    v4h h;
#pragma unroll
    for (int j = 0; j < 4; ++j) h[j] = (_Float16)fmaxf(nm * wv[j] + bv[j], 0.0f);
    *(v4h*)(&hbuf[wave][idx4]) = h;
  }
  LDS_FENCE();
  const int g = lane >> 4, cl = lane & 15;
#pragma unroll
  for (int ct = 0; ct < 4; ++ct) {
    v8f acc = {};
#pragma unroll
    for (int kc = 0; kc < 2; ++kc) {
      v16h a = lda_f16(&hbuf[wave][0], 64, kc * 32, lane);
      v16h b = ldb_packed(w2p, 4, kc, ct, lane);
      acc = WMMA_F16(a, b, acc);
    }
    const int col = ct * 16 + cl;
    const float bias = b2[col];       // no final relu in mlp2
    if (full) {
#pragma unroll
      for (int v = 0; v < 8; ++v) x0[(long)(rbase + v + g * 8) * 64 + col] = acc[v] + bias;
    } else {
#pragma unroll
      for (int v = 0; v < 8; ++v) {
        const int gr = rbase + v + g * 8;
        if (gr < N) x0[(long)gr * 64 + col] = acc[v] + bias;
      }
    }
  }
}

// ---- fused GINE message: e=mlp2(attr); m=relu(x[src]+e@lin+linb); agg[dst]+=m
template <int DIN>
__global__ void gine_message_kernel(const int* __restrict__ esrc, const int* __restrict__ edst,
                                    const float* __restrict__ eattr,
                                    const float* __restrict__ x,        // [N, DIN]
                                    const float* __restrict__ ew1, const float* __restrict__ eb1,
                                    const _Float16* __restrict__ ew2p,  // packed [64,64]
                                    const float* __restrict__ eb2,
                                    const _Float16* __restrict__ linp,  // packed [64,DIN]
                                    const float* __restrict__ linb,
                                    float* __restrict__ agg,            // [N, DIN]
                                    int E) {
  const int wave = threadIdx.x >> 5, lane = threadIdx.x & 31;
  const int tile = blockIdx.x * 4 + wave;
  const int numTiles = (E + 15) >> 4;
  __shared__ float    abuf[4][16];
  __shared__ int      sd[4][32];
  __shared__ alignas(16) _Float16 ubuf[4][16 * 64];
  __shared__ alignas(16) _Float16 ebuf[4][16 * 64];
  if (tile >= numTiles) return;
  const long ebase = (long)tile * 16;
  const bool full = (ebase + 16 <= E);
  if (lane < 16) {
    const long ei = ebase + lane;
    const bool ok = full || ei < E;
    abuf[wave][lane]    = ok ? eattr[ei] : 0.0f;
    sd[wave][lane]      = ok ? esrc[ei] : 0;
    sd[wave][16 + lane] = ok ? edst[ei] : 0;
  }
  LDS_FENCE();
  // edge-MLP hidden u = relu(attr * ew1 + eb1), 16x64
#pragma unroll
  for (int t = 0; t < 8; ++t) {
    const int idx4 = (lane + 32 * t) * 4;
    const int r = idx4 >> 6, c = idx4 & 63;
    const float a = abuf[wave][r];
    const v4f wv = *(const v4f*)(ew1 + c);
    const v4f bv = *(const v4f*)(eb1 + c);
    v4h h;
#pragma unroll
    for (int j = 0; j < 4; ++j) h[j] = (_Float16)fmaxf(a * wv[j] + bv[j], 0.0f);
    *(v4h*)(&ubuf[wave][idx4]) = h;
  }
  LDS_FENCE();
  const int g = lane >> 4, cl = lane & 15;
  // edge embedding e = u @ ew2 + eb2  (no relu), kept in LDS as f16
#pragma unroll
  for (int ct = 0; ct < 4; ++ct) {
    v8f acc = {};
#pragma unroll
    for (int kc = 0; kc < 2; ++kc) {
      v16h a = lda_f16(&ubuf[wave][0], 64, kc * 32, lane);
      v16h b = ldb_packed(ew2p, 4, kc, ct, lane);
      acc = WMMA_F16(a, b, acc);
    }
    const int col = ct * 16 + cl;
    const float bias = eb2[col];
#pragma unroll
    for (int v = 0; v < 8; ++v) {
      ebuf[wave][(v + g * 8) * 64 + col] = (_Float16)(acc[v] + bias);
    }
  }
  LDS_FENCE();
  // m = relu(x[src] + e @ lin + linb); scatter-add into agg[dst]
#pragma unroll
  for (int ct = 0; ct < DIN / 16; ++ct) {
    v8f acc = {};
#pragma unroll
    for (int kc = 0; kc < 2; ++kc) {
      v16h a = lda_f16(&ebuf[wave][0], 64, kc * 32, lane);
      v16h b = ldb_packed(linp, DIN / 16, kc, ct, lane);
      acc = WMMA_F16(a, b, acc);
    }
    const int col = ct * 16 + cl;
    const float lb = linb[col];
    if (full) {
#pragma unroll
      for (int v = 0; v < 8; ++v) {
        const int r = v + g * 8;
        const int s = sd[wave][r], d = sd[wave][16 + r];
        const float val = fmaxf(acc[v] + lb + x[(long)s * DIN + col], 0.0f);
        atomicAdd(&agg[(long)d * DIN + col], val);
      }
    } else {
#pragma unroll
      for (int v = 0; v < 8; ++v) {
        const int r = v + g * 8;
        if (ebase + r < E) {
          const int s = sd[wave][r], d = sd[wave][16 + r];
          const float val = fmaxf(acc[v] + lb + x[(long)s * DIN + col], 0.0f);
          atomicAdd(&agg[(long)d * DIN + col], val);
        }
      }
    }
  }
}

// ---- GINE update: out = relu(mlp2(x+agg, w1[DIN,128], w2[128,128])) --------
template <int DIN>
__global__ void gine_update_kernel(const float* __restrict__ x, const float* __restrict__ agg,
                                   const _Float16* __restrict__ w1p, const float* __restrict__ b1,
                                   const _Float16* __restrict__ w2p, const float* __restrict__ b2,
                                   float* __restrict__ out,  // [N,128]
                                   int N) {
  const int wave = threadIdx.x >> 5, lane = threadIdx.x & 31;
  const int tile = blockIdx.x * 4 + wave;
  const int numTiles = (N + 15) >> 4;
  __shared__ alignas(16) _Float16 a0[4][16 * DIN];
  __shared__ alignas(16) _Float16 hb[4][16 * 128];
  if (tile >= numTiles) return;
  const int rbase = tile * 16;
  const bool full = (rbase + 16 <= N);
#pragma unroll
  for (int t = 0; t < (16 * DIN) / 128; ++t) {
    const int idx4 = (lane + 32 * t) * 4;
    const int r = idx4 / DIN, c = idx4 % DIN;
    const int gr = rbase + r;
    v4h h;
    if (full || gr < N) {
      const v4f xv = *(const v4f*)(x + (long)gr * DIN + c);
      const v4f av = *(const v4f*)(agg + (long)gr * DIN + c);
#pragma unroll
      for (int j = 0; j < 4; ++j) h[j] = (_Float16)(xv[j] + av[j]);
    } else {
#pragma unroll
      for (int j = 0; j < 4; ++j) h[j] = (_Float16)0.0f;
    }
    *(v4h*)(&a0[wave][idx4]) = h;
  }
  LDS_FENCE();
  const int g = lane >> 4, cl = lane & 15;
#pragma unroll
  for (int ct = 0; ct < 8; ++ct) {
    v8f acc = {};
#pragma unroll
    for (int kc = 0; kc < DIN / 32; ++kc) {
      v16h a = lda_f16(&a0[wave][0], DIN, kc * 32, lane);
      v16h b = ldb_packed(w1p, 8, kc, ct, lane);
      acc = WMMA_F16(a, b, acc);
    }
    const int col = ct * 16 + cl;
    const float bias = b1[col];
#pragma unroll
    for (int v = 0; v < 8; ++v) {
      hb[wave][(v + g * 8) * 128 + col] = (_Float16)fmaxf(acc[v] + bias, 0.0f);
    }
  }
  LDS_FENCE();
#pragma unroll
  for (int ct = 0; ct < 8; ++ct) {
    v8f acc = {};
#pragma unroll
    for (int kc = 0; kc < 4; ++kc) {
      v16h a = lda_f16(&hb[wave][0], 128, kc * 32, lane);
      v16h b = ldb_packed(w2p, 8, kc, ct, lane);
      acc = WMMA_F16(a, b, acc);
    }
    const int col = ct * 16 + cl;
    const float bias = b2[col];
    if (full) {
#pragma unroll
      for (int v = 0; v < 8; ++v)
        out[(long)(rbase + v + g * 8) * 128 + col] = fmaxf(acc[v] + bias, 0.0f);  // outer relu
    } else {
#pragma unroll
      for (int v = 0; v < 8; ++v) {
        const int gr = rbase + v + g * 8;
        if (gr < N) out[(long)gr * 128 + col] = fmaxf(acc[v] + bias, 0.0f);
      }
    }
  }
}

// ---- mean-pool helpers -----------------------------------------------------
__global__ void pool_sum_kernel(const float* __restrict__ x, const int* __restrict__ batch,
                                float* __restrict__ sums, long n /* N*128 */) {
  long i = (long)blockIdx.x * blockDim.x + threadIdx.x;
  if (i < n) {
    const int node = (int)(i >> 7), c = (int)(i & 127);
    atomicAdd(&sums[(long)batch[node] * 128 + c], x[i]);
  }
}
__global__ void pool_cnt_kernel(const int* __restrict__ batch, float* __restrict__ cnt, int N) {
  int i = blockIdx.x * blockDim.x + threadIdx.x;
  if (i < N) atomicAdd(&cnt[batch[i]], 1.0f);
}

// ---- regressor head --------------------------------------------------------
__global__ void final_kernel(const float* __restrict__ sums_s, const float* __restrict__ cnt_s,
                             const float* __restrict__ sums_g, const float* __restrict__ cnt_g,
                             const float* __restrict__ depth,
                             const float* __restrict__ w1, const float* __restrict__ b1,
                             const float* __restrict__ w2, const float* __restrict__ b2,
                             float* __restrict__ out, int B) {
  const int b = blockIdx.x;
  const int c = threadIdx.x;  // 0..127
  __shared__ float zs[128], zg[128], red[128];
  const float invs = 1.0f / fmaxf(cnt_s[b], 1.0f);
  const float invg = 1.0f / fmaxf(cnt_g[b], 1.0f);
  zs[c] = sums_s[b * 128 + c] * invs;
  zg[c] = sums_g[b * 128 + c] * invg;
  __syncthreads();
  float acc = b1[c];
  for (int k = 0; k < 128; ++k) acc += zs[k] * w1[k * 128 + c];
  for (int k = 0; k < 128; ++k) acc += zg[k] * w1[(128 + k) * 128 + c];
  acc += depth[b] * w1[256 * 128 + c];
  red[c] = fmaxf(acc, 0.0f) * w2[c];
  __syncthreads();
  for (int s = 64; s > 0; s >>= 1) {
    if (c < s) red[c] += red[c + s];
    __syncthreads();
  }
  if (c == 0 && b < B) out[b] = red[0] + b2[0];
}

// ---------------------------------------------------------------------------
extern "C" void kernel_launch(void* const* d_in, const int* in_sizes, int n_in,
                              void* d_out, int out_size, void* d_ws, size_t ws_size,
                              hipStream_t stream) {
  const int N = in_sizes[0];
  const int E = in_sizes[2];
  const int B = in_sizes[8];
  auto P = [&](int i) -> const float* { return (const float*)d_in[i]; };

  char* wsb = (char*)d_ws;
  size_t off = 0;
  auto alloc = [&](size_t bytes) -> char* {
    off = (off + 255) & ~(size_t)255;
    char* p = wsb + off;
    off += bytes;
    return p;
  };

  // fragment-packed f16 weight staging
  _Float16* id_w2p = (_Float16*)alloc(64 * 64 * 2);
  _Float16* ed_w2p = (_Float16*)alloc(64 * 64 * 2);
  _Float16 *l1_lin[2], *l1_w1[2], *l1_w2[2], *l2_lin[2], *l2_w1[2], *l2_w2[2];
  for (int e = 0; e < 2; ++e) {
    l1_lin[e] = (_Float16*)alloc(64 * 64 * 2);
    l1_w1[e]  = (_Float16*)alloc(64 * 128 * 2);
    l1_w2[e]  = (_Float16*)alloc(128 * 128 * 2);
    l2_lin[e] = (_Float16*)alloc(64 * 128 * 2);
    l2_w1[e]  = (_Float16*)alloc(128 * 128 * 2);
    l2_w2[e]  = (_Float16*)alloc(128 * 128 * 2);
  }
  // activation scratch
  float* x0   = (float*)alloc((size_t)N * 64 * 4);
  float* x1   = (float*)alloc((size_t)N * 128 * 4);
  float* aggb = (float*)alloc((size_t)N * 128 * 4);
  float* x2   = (float*)alloc((size_t)N * 128 * 4);
  float* sums[2] = {(float*)alloc((size_t)B * 128 * 4), (float*)alloc((size_t)B * 128 * 4)};
  float* cnts[2] = {(float*)alloc((size_t)B * 4), (float*)alloc((size_t)B * 4)};

  auto pack = [&](const float* s, _Float16* d, int K, int Ncols) {
    const int n = K * Ncols;
    pack_b_kernel<<<(n + 255) / 256, 256, 0, stream>>>(s, d, K, Ncols);
  };
  // params: 9 id_w1,10 id_b1,11 id_w2,12 id_b2, 13 ed_w1,14 ed_b1,15 ed_w2,16 ed_b2,
  //         17 reg_w1,18 reg_b1,19 reg_w2,20 reg_b2, then s1@21 s2@27 g1@33 g2@39
  pack(P(11), id_w2p, 64, 64);
  pack(P(15), ed_w2p, 64, 64);
  for (int e = 0; e < 2; ++e) {
    const int b1i = e ? 33 : 21, b2i = e ? 39 : 27;
    pack(P(b1i + 0), l1_lin[e], 64, 64);
    pack(P(b1i + 2), l1_w1[e], 64, 128);
    pack(P(b1i + 4), l1_w2[e], 128, 128);
    pack(P(b2i + 0), l2_lin[e], 64, 128);
    pack(P(b2i + 2), l2_w1[e], 128, 128);
    pack(P(b2i + 4), l2_w2[e], 128, 128);
  }

  const int ntiles = (N + 15) / 16, nblocks = (ntiles + 3) / 4;
  const int etiles = (E + 15) / 16, eblocks = (etiles + 3) / 4;

  for (int enc = 0; enc < 2; ++enc) {
    const int*   names = (const int*)d_in[enc ? 4 : 0];
    const int*   eidx  = (const int*)d_in[enc ? 5 : 1];
    const float* eattr = (const float*)d_in[enc ? 6 : 2];
    const int*   batch = (const int*)d_in[enc ? 7 : 3];
    const int*   esrc = eidx;
    const int*   edst = eidx + E;
    const int    b1i = enc ? 33 : 21, b2i = enc ? 39 : 27;

    node_encode_kernel<<<nblocks, 128, 0, stream>>>(names, P(9), P(10), id_w2p, P(12), x0, N);

    zero_kernel<<<(int)(((long)N * 64 + 255) / 256), 256, 0, stream>>>(aggb, (long)N * 64);
    gine_message_kernel<64><<<eblocks, 128, 0, stream>>>(
        esrc, edst, eattr, x0, P(13), P(14), ed_w2p, P(16), l1_lin[enc], P(b1i + 1), aggb, E);
    gine_update_kernel<64><<<nblocks, 128, 0, stream>>>(
        x0, aggb, l1_w1[enc], P(b1i + 3), l1_w2[enc], P(b1i + 5), x1, N);

    zero_kernel<<<(int)(((long)N * 128 + 255) / 256), 256, 0, stream>>>(aggb, (long)N * 128);
    gine_message_kernel<128><<<eblocks, 128, 0, stream>>>(
        esrc, edst, eattr, x1, P(13), P(14), ed_w2p, P(16), l2_lin[enc], P(b2i + 1), aggb, E);
    gine_update_kernel<128><<<nblocks, 128, 0, stream>>>(
        x1, aggb, l2_w1[enc], P(b2i + 3), l2_w2[enc], P(b2i + 5), x2, N);

    zero_kernel<<<(B * 128 + 255) / 256, 256, 0, stream>>>(sums[enc], (long)B * 128);
    zero_kernel<<<(B + 255) / 256, 256, 0, stream>>>(cnts[enc], (long)B);
    pool_sum_kernel<<<(int)(((long)N * 128 + 255) / 256), 256, 0, stream>>>(
        x2, batch, sums[enc], (long)N * 128);
    pool_cnt_kernel<<<(N + 255) / 256, 256, 0, stream>>>(batch, cnts[enc], N);
  }

  final_kernel<<<B, 128, 0, stream>>>(sums[0], cnts[0], sums[1], cnts[1],
                                      (const float*)d_in[8], P(17), P(18), P(19), P(20),
                                      (float*)d_out, B);
}
